// GATGraphClassifier_29884382446362
// MI455X (gfx1250) — matrix operand order repaired and checked by reading, not compiled, not run on hardware
//
#include <hip/hip_runtime.h>

#define N_NODES 50000
#define N_EDGES 800000
#define IN_DIM 128
#define HIDDEN 64
#define HEADS 4
#define NUM_CLASSES 2
#define NUM_GRAPHS 500
#define NEG_SLOPE 0.2f
#define ET (N_EDGES + N_NODES)   // edges + self loops

typedef __attribute__((ext_vector_type(16))) __bf16 v16bf;
typedef __attribute__((ext_vector_type(8)))  float v8f;
typedef __attribute__((ext_vector_type(8)))  unsigned short v8u;
typedef __attribute__((ext_vector_type(16))) unsigned short v16u;

// ---------- helpers ----------

__device__ __forceinline__ unsigned short f2bf(float f) {
  unsigned u = __float_as_uint(f);
  u += 0x7fffu + ((u >> 16) & 1u);            // round-to-nearest-even
  return (unsigned short)(u >> 16);
}

__device__ __forceinline__ float lrelu(float v) {
  return v > 0.f ? v : NEG_SLOPE * v;
}

__device__ __forceinline__ void atomicMaxF(float* addr, float v) {
  if (v >= 0.f) atomicMax((int*)addr, __float_as_int(v));
  else          atomicMin((unsigned int*)addr, __float_as_uint(v));
}

__device__ __forceinline__ void edge_sd(const long long* __restrict__ ei,
                                        long long e, int& s, int& d) {
  if (e < (long long)N_EDGES) {
    s = (int)ei[e];
    d = (int)ei[(long long)N_EDGES + e];
  } else {
    s = d = (int)(e - N_EDGES);               // self loop
  }
}

// ---------- generic utility kernels ----------

__global__ void k_fill(float* __restrict__ p, float v, long long n) {
  long long i = (long long)blockIdx.x * blockDim.x + threadIdx.x;
  if (i < n) p[i] = v;
}

__global__ void k_cvt_bf16(const float* __restrict__ in,
                           unsigned short* __restrict__ out, long long n) {
  long long i = (long long)blockIdx.x * blockDim.x + threadIdx.x;
  if (i < n) out[i] = f2bf(in[i]);
}

// Pack W[K,Nn] (f32 row-major) into per-wave WMMA B fragments (bf16).
// Fragment f = ntile*Kb + kb holds 32 lanes x 16 elems, lane-contiguous.
// Element (lane,e): n = ntile*16 + (lane&15), k = kb*32 + (lane>>4)*16 + e.
__global__ void k_packB(const float* __restrict__ W,
                        unsigned short* __restrict__ pack, int K, int Nn) {
  int Kb = K >> 5;
  long long total = (long long)Kb * (Nn >> 4) * 512;
  long long idx = (long long)blockIdx.x * blockDim.x + threadIdx.x;
  if (idx >= total) return;
  int f    = (int)(idx >> 9);
  int r    = (int)(idx & 511);
  int lane = r >> 4;
  int e    = r & 15;
  int ntile = f / Kb;
  int kb    = f % Kb;
  int n = ntile * 16 + (lane & 15);
  int k = kb * 32 + (lane >> 4) * 16 + e;
  pack[idx] = f2bf(W[(long long)k * Nn + n]);
}

// ---------- WMMA GEMM: C[M,Nn] = A[M,K]_bf16 @ Bpack(K,Nn)_bf16 ----------
// blockIdx.y = ntile; blockIdx.x*4 + wave = mtile. 4 waves (128 thr) / block.
// All waves of a block share one ntile -> stage its packed B column in LDS
// via global_load_async_to_lds_b128 (ASYNCcnt path), then feed WMMA from DS.
__global__ void k_gemm_bf16(const unsigned short* __restrict__ A,
                            const unsigned short* __restrict__ Bp,
                            float* __restrict__ C,
                            int M, int K, int Nn) {
  __shared__ __align__(16) unsigned short ldsB[8 * 512];  // up to Kb=8 (8 KB)
  int Mt = M >> 4, Kb = K >> 5;
  int ntile = blockIdx.y;
  int mtile = blockIdx.x * 4 + (threadIdx.x >> 5);
  int lane = threadIdx.x & 31;
  int half = lane >> 4;
  int mr   = lane & 15;

  // async-copy this ntile's packed B (Kb*1024 bytes) into LDS
  const unsigned short* bsrc = Bp + (((size_t)ntile * Kb) << 9);
  unsigned ldsBase = (unsigned)(uintptr_t)&ldsB[0];   // low 32 bits = LDS offset
  int packBytes = Kb << 10;
  for (int ofs = threadIdx.x * 16; ofs < packBytes; ofs += 128 * 16) {
    unsigned la = ldsBase + (unsigned)ofs;
    unsigned long long ga = (unsigned long long)(uintptr_t)bsrc + (unsigned)ofs;
    asm volatile("global_load_async_to_lds_b128 %0, %1, off"
                 :: "v"(la), "v"(ga) : "memory");
  }
  asm volatile("s_wait_asynccnt 0x0" ::: "memory");
  __syncthreads();

  bool valid = mtile < Mt;
  int row = valid ? (mtile * 16 + mr) : mr;           // clamp to safe row
  const unsigned short* ap0 = A + (size_t)row * K + half * 16;

  v8f acc = {};
  for (int kb = 0; kb < Kb; ++kb) {
    const unsigned short* ap = ap0 + kb * 32;
    v8u alo = *(const v8u*)ap;                        // global_load_b128
    v8u ahi = *(const v8u*)(ap + 8);
    const unsigned short* bp = ldsB + ((size_t)kb << 9) + lane * 16;
    v8u blo = *(const v8u*)bp;                        // ds_load_b128
    v8u bhi = *(const v8u*)(bp + 8);
    if (kb + 1 < Kb) __builtin_prefetch(ap + 32, 0, 3);  // global_prefetch_b8

    v16u au, bu;
#pragma unroll
    for (int i = 0; i < 8; ++i) {
      au[i] = alo[i]; au[i + 8] = ahi[i];
      bu[i] = blo[i]; bu[i + 8] = bhi[i];
    }
    acc = __builtin_amdgcn_wmma_f32_16x16x32_bf16(
        false, __builtin_bit_cast(v16bf, au),
        false, __builtin_bit_cast(v16bf, bu),
        (short)0, acc, false, false);
  }

  if (!valid) return;
  // C/D layout: VGPR r -> row mtile*16 + half*8 + r, col ntile*16 + mr
  float* cp = C + (size_t)(mtile * 16 + half * 8) * Nn + ntile * 16 + mr;
#pragma unroll
  for (int r = 0; r < 8; ++r) cp[(size_t)r * Nn] = acc[r];
}

// ---------- attention coefficients: as[n,h] = <h[n,h,:], a_src[h,:]> ----------
__global__ void k_attn(const float* __restrict__ h,
                       const float* __restrict__ asrc,
                       const float* __restrict__ adst,
                       float* __restrict__ as_, float* __restrict__ ad_,
                       int N, int H, int C) {
  long long idx = (long long)blockIdx.x * blockDim.x + threadIdx.x;
  if (idx >= (long long)N * H) return;
  int hh = (int)(idx % H);
  long long n = idx / H;
  const float* hp = h + (n * H + hh) * C;
  const float* sp = asrc + (long long)hh * C;
  const float* dp = adst + (long long)hh * C;
  float vs = 0.f, vd = 0.f;
  for (int c = 0; c < C; ++c) {
    float x = hp[c];
    vs += x * sp[c];
    vd += x * dp[c];
  }
  as_[idx] = vs;
  ad_[idx] = vd;
}

// ---------- edge passes ----------

__global__ void k_edge_max(const long long* __restrict__ ei,
                           const float* __restrict__ as_,
                           const float* __restrict__ ad_,
                           float* __restrict__ m, int H) {
  long long idx = (long long)blockIdx.x * blockDim.x + threadIdx.x;
  if (idx >= (long long)ET * H) return;
  long long e = idx / H;
  int hh = (int)(idx % H);
  int s, d;
  edge_sd(ei, e, s, d);
  float v = lrelu(as_[(long long)s * H + hh] + ad_[(long long)d * H + hh]);
  atomicMaxF(&m[(long long)d * H + hh], v);
}

__global__ void k_edge_expsum(const long long* __restrict__ ei,
                              const float* __restrict__ as_,
                              const float* __restrict__ ad_,
                              const float* __restrict__ m,
                              float* __restrict__ den, int H) {
  long long idx = (long long)blockIdx.x * blockDim.x + threadIdx.x;
  if (idx >= (long long)ET * H) return;
  long long e = idx / H;
  int hh = (int)(idx % H);
  int s, d;
  edge_sd(ei, e, s, d);
  float v = lrelu(as_[(long long)s * H + hh] + ad_[(long long)d * H + hh]);
  float ex = __expf(v - m[(long long)d * H + hh]);
  atomicAdd(&den[(long long)d * H + hh], ex);
}

__global__ void k_edge_aggr(const long long* __restrict__ ei,
                            const float* __restrict__ as_,
                            const float* __restrict__ ad_,
                            const float* __restrict__ m,
                            const float* __restrict__ den,
                            const float* __restrict__ h,
                            float* __restrict__ out, int H, int C) {
  long long idx = (long long)blockIdx.x * blockDim.x + threadIdx.x;
  int F = H * C;
  if (idx >= (long long)ET * F) return;
  long long e = idx / F;
  int r = (int)(idx % F);
  int hh = r / C;
  int s, d;
  edge_sd(ei, e, s, d);
  float v = lrelu(as_[(long long)s * H + hh] + ad_[(long long)d * H + hh]);
  float ex = __expf(v - m[(long long)d * H + hh]);
  float alpha = ex / (den[(long long)d * H + hh] + 1e-16f);
  atomicAdd(&out[(long long)d * F + r], alpha * h[(long long)s * F + r]);
}

// ---------- bias + relu + bf16 convert ----------
__global__ void k_bias_relu_cvt(const float* __restrict__ in,
                                const float* __restrict__ bias,
                                unsigned short* __restrict__ out,
                                long long n, int F) {
  long long i = (long long)blockIdx.x * blockDim.x + threadIdx.x;
  if (i >= n) return;
  float v = in[i] + bias[(int)(i % F)];
  v = fmaxf(v, 0.f);
  out[i] = f2bf(v);
}

// ---------- mean pool (accumulate) ----------
__global__ void k_pool(const float* __restrict__ in, const float* __restrict__ bias,
                       const long long* __restrict__ batch,
                       float* __restrict__ sums, float* __restrict__ cnt,
                       int N, int F) {
  long long idx = (long long)blockIdx.x * blockDim.x + threadIdx.x;
  if (idx >= (long long)N * F) return;
  long long n = idx / F;
  int c = (int)(idx % F);
  float v = fmaxf(in[idx] + bias[c], 0.f);
  int g = (int)batch[n];
  atomicAdd(&sums[(long long)g * F + c], v);
  if (c == 0) atomicAdd(&cnt[g], 1.0f);
}

// ---------- classifier ----------
__global__ void k_cls(const float* __restrict__ sums, const float* __restrict__ cnt,
                      const float* __restrict__ Wl, const float* __restrict__ bl,
                      float* __restrict__ out, int G, int F, int NC) {
  int idx = blockIdx.x * blockDim.x + threadIdx.x;
  if (idx >= G * NC) return;
  int g = idx / NC, k = idx % NC;
  float c = fmaxf(cnt[g], 1.0f);
  float s = bl[k];
  for (int f = 0; f < F; ++f) s += (sums[(long long)g * F + f] / c) * Wl[f * NC + k];
  out[idx] = s;
}

// ---------- launch ----------

extern "C" void kernel_launch(void* const* d_in, const int* in_sizes, int n_in,
                              void* d_out, int out_size, void* d_ws, size_t ws_size,
                              hipStream_t stream) {
  (void)in_sizes; (void)n_in; (void)out_size; (void)ws_size;
  const float*     x      = (const float*)d_in[0];
  const long long* ei     = (const long long*)d_in[1];
  const long long* batch  = (const long long*)d_in[3];
  const float*     W1     = (const float*)d_in[4];
  const float*     a_src1 = (const float*)d_in[5];
  const float*     a_dst1 = (const float*)d_in[6];
  const float*     b1     = (const float*)d_in[7];
  const float*     W2     = (const float*)d_in[8];
  const float*     a_src2 = (const float*)d_in[9];
  const float*     a_dst2 = (const float*)d_in[10];
  const float*     b2     = (const float*)d_in[11];
  const float*     W_lin  = (const float*)d_in[12];
  const float*     b_lin  = (const float*)d_in[13];
  float* out = (float*)d_out;
  char*  ws  = (char*)d_ws;

  // workspace layout (bytes); regions reused once dead
  const size_t OFF_H1RAW  = 0;           // f32 [50000,256]  51,200,000
  const size_t OFF_H1ACTB = 0;           // bf16 [50000,256] 25,600,000 (aliases H1RAW)
  const size_t OFF_OUT2   = 25600000;    // f32 [50000,64]   12,800,000 (aliases H1RAW tail)
  const size_t OFF_OUT1   = 51200000;    // f32 [50000,256]  51,200,000
  const size_t OFF_XB     = 102400000;   // bf16 [50000,128] 12,800,000
  const size_t OFF_H2RAW  = OFF_XB;      // f32 [50000,64]   (aliases XB)
  const size_t OFF_W1P    = 115200000;   // 65,536
  const size_t OFF_W2P    = 115265536;   // 32,768
  const size_t OFF_AS1    = 115298304;   // 800,000
  const size_t OFF_AD1    = 116098304;
  const size_t OFF_M1     = 116898304;
  const size_t OFF_D1     = 117698304;
  const size_t OFF_AS2    = 118498304;   // 200,000
  const size_t OFF_AD2    = 118698304;
  const size_t OFF_M2     = 118898304;
  const size_t OFF_D2     = 119098304;
  const size_t OFF_PS     = 119298304;   // 128,000
  const size_t OFF_PC     = 119426304;   // 2,000

  auto F32 = [&](size_t off) { return (float*)(ws + off); };
  auto U16 = [&](size_t off) { return (unsigned short*)(ws + off); };
  auto nb  = [](long long n, int t) { return dim3((unsigned)((n + t - 1) / t)); };

  const long long NH1 = (long long)N_NODES * HEADS;                 // 200000
  const long long NF1 = (long long)N_NODES * HEADS * HIDDEN;        // 12.8M
  const long long NF2 = (long long)N_NODES * HIDDEN;                // 3.2M

  // ---- layer 1 ----
  k_cvt_bf16<<<nb((long long)N_NODES * IN_DIM, 256), 256, 0, stream>>>(
      x, U16(OFF_XB), (long long)N_NODES * IN_DIM);
  k_packB<<<nb(4 * 16 * 512, 256), 256, 0, stream>>>(W1, U16(OFF_W1P), IN_DIM, HEADS * HIDDEN);
  k_packB<<<nb(8 * 4 * 512, 256), 256, 0, stream>>>(W2, U16(OFF_W2P), HEADS * HIDDEN, HIDDEN);

  // GEMM1: grid = (ceil(3125/4)=782 mtile-groups, 16 ntiles), 4 waves/block
  k_gemm_bf16<<<dim3(782, 16), 128, 0, stream>>>(U16(OFF_XB), U16(OFF_W1P), F32(OFF_H1RAW),
                                                 N_NODES, IN_DIM, HEADS * HIDDEN);

  k_attn<<<nb(NH1, 256), 256, 0, stream>>>(F32(OFF_H1RAW), a_src1, a_dst1,
                                           F32(OFF_AS1), F32(OFF_AD1),
                                           N_NODES, HEADS, HIDDEN);
  k_fill<<<nb(NH1, 256), 256, 0, stream>>>(F32(OFF_M1), -1e30f, NH1);
  k_fill<<<nb(NH1, 256), 256, 0, stream>>>(F32(OFF_D1), 0.f, NH1);
  k_fill<<<nb(NF1, 256), 256, 0, stream>>>(F32(OFF_OUT1), 0.f, NF1);

  k_edge_max<<<nb((long long)ET * HEADS, 256), 256, 0, stream>>>(
      ei, F32(OFF_AS1), F32(OFF_AD1), F32(OFF_M1), HEADS);
  k_edge_expsum<<<nb((long long)ET * HEADS, 256), 256, 0, stream>>>(
      ei, F32(OFF_AS1), F32(OFF_AD1), F32(OFF_M1), F32(OFF_D1), HEADS);
  k_edge_aggr<<<nb((long long)ET * HEADS * HIDDEN, 256), 256, 0, stream>>>(
      ei, F32(OFF_AS1), F32(OFF_AD1), F32(OFF_M1), F32(OFF_D1),
      F32(OFF_H1RAW), F32(OFF_OUT1), HEADS, HIDDEN);

  k_bias_relu_cvt<<<nb(NF1, 256), 256, 0, stream>>>(F32(OFF_OUT1), b1,
                                                    U16(OFF_H1ACTB), NF1, HEADS * HIDDEN);

  // ---- layer 2 ----
  // GEMM2: grid = (782 mtile-groups, 4 ntiles)
  k_gemm_bf16<<<dim3(782, 4), 128, 0, stream>>>(U16(OFF_H1ACTB), U16(OFF_W2P), F32(OFF_H2RAW),
                                                N_NODES, HEADS * HIDDEN, HIDDEN);

  k_attn<<<nb(N_NODES, 256), 256, 0, stream>>>(F32(OFF_H2RAW), a_src2, a_dst2,
                                               F32(OFF_AS2), F32(OFF_AD2),
                                               N_NODES, 1, HIDDEN);
  k_fill<<<nb(N_NODES, 256), 256, 0, stream>>>(F32(OFF_M2), -1e30f, N_NODES);
  k_fill<<<nb(N_NODES, 256), 256, 0, stream>>>(F32(OFF_D2), 0.f, N_NODES);
  k_fill<<<nb(NF2, 256), 256, 0, stream>>>(F32(OFF_OUT2), 0.f, NF2);

  k_edge_max<<<nb((long long)ET, 256), 256, 0, stream>>>(
      ei, F32(OFF_AS2), F32(OFF_AD2), F32(OFF_M2), 1);
  k_edge_expsum<<<nb((long long)ET, 256), 256, 0, stream>>>(
      ei, F32(OFF_AS2), F32(OFF_AD2), F32(OFF_M2), F32(OFF_D2), 1);
  k_edge_aggr<<<nb((long long)ET * HIDDEN, 256), 256, 0, stream>>>(
      ei, F32(OFF_AS2), F32(OFF_AD2), F32(OFF_M2), F32(OFF_D2),
      F32(OFF_H2RAW), F32(OFF_OUT2), 1, HIDDEN);

  // ---- pool + classifier ----
  k_fill<<<nb((long long)NUM_GRAPHS * HIDDEN, 256), 256, 0, stream>>>(
      F32(OFF_PS), 0.f, (long long)NUM_GRAPHS * HIDDEN);
  k_fill<<<nb(NUM_GRAPHS, 256), 256, 0, stream>>>(F32(OFF_PC), 0.f, NUM_GRAPHS);
  k_pool<<<nb(NF2, 256), 256, 0, stream>>>(F32(OFF_OUT2), b2, batch,
                                           F32(OFF_PS), F32(OFF_PC), N_NODES, HIDDEN);
  k_cls<<<nb(NUM_GRAPHS * NUM_CLASSES, 256), 256, 0, stream>>>(
      F32(OFF_PS), F32(OFF_PC), W_lin, b_lin, out, NUM_GRAPHS, HIDDEN, NUM_CLASSES);
}
